// EGKN_32452772888798
// MI455X (gfx1250) — matrix operand order repaired and checked by reading, not compiled, not run on hardware
//
#include <hip/hip_runtime.h>
#include <hip/hip_bf16.h>

// ---------------------------------------------------------------------------
// EGKN for MI455X (gfx1250, wave32, WMMA + TDM).
// Dominant cost: per-edge kernel GEMM k[E,128] @ W3[128,1024] fused with the
// eij,ej->ei contraction so the 410MB/layer Wk tensor never hits HBM.
// W3 (f16, 256KB) is staged into LDS by the Tensor Data Mover and shared by
// all waves in the block, double-buffered against the WMMA pipeline.
// ---------------------------------------------------------------------------

typedef _Float16 h8  __attribute__((ext_vector_type(8)));
typedef _Float16 h16 __attribute__((ext_vector_type(16)));
typedef float    v8f __attribute__((ext_vector_type(8)));

#define EGKN_N 10000
#define EGKN_E 100000
#define FUSED_WAVES 4     // waves (16-edge tiles) per block
#define GSZ 4             // W3 N-chunks per TDM stage group (4 * 4KB = 16KB)
#define NGROUPS (64 / GSZ)

// ---------------------------------------------------------------------------
// TDM: 1-D tensor DMA global->LDS, `ndw` dwords (must fit tile_dim0 16 bits).
// D# group0: count=1, lds_addr, 57b global addr, type=2.
// D# group1: data_size=4B, tensor_dim0=tile_dim0=ndw, tensor_dim1=tile_dim1=1,
//            tensor_dim0_stride=ndw.
// This toolchain's builtin takes 6 args (extra int32x8 + cpol).
// ---------------------------------------------------------------------------
__device__ __forceinline__ void tdm_load_1d(const void* gptr, unsigned lds_off,
                                            unsigned ndw) {
  typedef unsigned int v4u __attribute__((ext_vector_type(4)));
  typedef int v8i __attribute__((ext_vector_type(8)));
  typedef int v4i __attribute__((ext_vector_type(4)));
  unsigned long long ga = (unsigned long long)(uintptr_t)gptr;
  v4u g0;
  g0[0] = 1u;                                   // count=1, user mode
  g0[1] = lds_off;                              // LDS byte address
  g0[2] = (unsigned)ga;                         // global_addr[31:0]
  g0[3] = (unsigned)((ga >> 32) & 0x01FFFFFFu)  // global_addr[56:32]
          | 0x80000000u;                        // type=2 ("image")
  v8i g1;
  g1[0] = (int)(2u << 16);                      // data_size = 4 bytes
  g1[1] = (int)((ndw & 0xFFFFu) << 16);         // tensor_dim0[15:0]
  g1[2] = (int)((ndw >> 16) & 0xFFFFu) | (1 << 16); // dim0[31:16] | dim1 lo = 1
  g1[3] = (int)((ndw & 0xFFFFu) << 16);         // dim1 hi = 0 | tile_dim0
  g1[4] = 1;                                    // tile_dim1 = 1, tile_dim2 = 0
  g1[5] = (int)ndw;                             // tensor_dim0_stride[31:0]
  g1[6] = 0;                                    // stride hi | dim1_stride lo
  g1[7] = 0;
  v4i z4 = {0, 0, 0, 0};
  v8i z8 = {0, 0, 0, 0, 0, 0, 0, 0};
  __builtin_amdgcn_tensor_load_to_lds(g0, g1, z4, z4, z8, 0);
}

// ---------------------------------------------------------------------------
// Weight prep: f32 -> f16, pre-swizzled into WMMA B-fragment order.
// B 16-bit 32x16 layout: half h of lane l = B[K = 32*kf + (h/8)*16 + (l/16)*8 + h%8][N0 + l%16]
// Stored so each (chunk,kf,lane) owns 16 contiguous halves (32B -> 2x b128).
// ---------------------------------------------------------------------------
__global__ void egkn_prep(const float* __restrict__ w2,   // [64,128]
                          const float* __restrict__ w3,   // [128,1024]
                          _Float16* __restrict__ w2f,     // [8][2][32][16]
                          _Float16* __restrict__ w3f) {   // [64][4][32][16]
  int t = blockIdx.x * blockDim.x + threadIdx.x;
  if (t < 8192) {
    int hh = t & 15, lane = (t >> 4) & 31, kf = (t >> 9) & 1, nc = t >> 10;
    int K = kf * 32 + (hh >> 3) * 16 + (lane >> 4) * 8 + (hh & 7);
    int Nn = nc * 16 + (lane & 15);
    w2f[t] = (_Float16)w2[K * 128 + Nn];
  }
  if (t < 131072) {
    int hh = t & 15, lane = (t >> 4) & 31, kf = (t >> 9) & 3, c = t >> 11;
    int K = kf * 32 + (hh >> 3) * 16 + (lane >> 4) * 8 + (hh & 7);
    int Nn = c * 16 + (lane & 15);
    w3f[t] = (_Float16)w3[K * 1024 + Nn];
  }
}

// h = x @ fc1_w + fc1_b ; coord = coords_init
__global__ void egkn_init(const float* __restrict__ x, const float* __restrict__ w,
                          const float* __restrict__ b, const float* __restrict__ ci,
                          float* __restrict__ h, float* __restrict__ coord, int n_) {
  int n = blockIdx.x * blockDim.x + threadIdx.x;
  if (n >= n_) return;
  float x0 = x[n * 3 + 0], x1 = x[n * 3 + 1], x2 = x[n * 3 + 2];
  #pragma unroll
  for (int d = 0; d < 32; ++d)
    h[(size_t)n * 32 + d] = b[d] + x0 * w[d] + x1 * w[32 + d] + x2 * w[64 + d];
  coord[n * 3 + 0] = ci[n * 3 + 0];
  coord[n * 3 + 1] = ci[n * 3 + 1];
  coord[n * 3 + 2] = ci[n * 3 + 2];
}

// Per-edge: coord_diff, radial, k1 = relu([edge_attr,radial] @ W1 + b1) -> f16
__global__ void egkn_edge1(const int* __restrict__ eidx, const float* __restrict__ ea,
                           const float* __restrict__ coord, const float* __restrict__ w1,
                           const float* __restrict__ b1, float* __restrict__ cd,
                           _Float16* __restrict__ k1, int e_) {
  int e = blockIdx.x * blockDim.x + threadIdx.x;
  if (e >= e_) return;
  int r = eidx[e], c = eidx[e_ + e];
  float d0 = coord[r * 3 + 0] - coord[c * 3 + 0];
  float d1 = coord[r * 3 + 1] - coord[c * 3 + 1];
  float d2 = coord[r * 3 + 2] - coord[c * 3 + 2];
  cd[(size_t)e * 3 + 0] = d0; cd[(size_t)e * 3 + 1] = d1; cd[(size_t)e * 3 + 2] = d2;
  float kin[7];
  #pragma unroll
  for (int i = 0; i < 6; ++i) kin[i] = ea[(size_t)e * 6 + i];
  kin[6] = d0 * d0 + d1 * d1 + d2 * d2;
  #pragma unroll 4
  for (int d = 0; d < 64; ++d) {
    float s = b1[d];
    #pragma unroll
    for (int i = 0; i < 7; ++i) s += kin[i] * w1[i * 64 + d];
    k1[(size_t)e * 64 + d] = (_Float16)(s > 0.f ? s : 0.f);
  }
}

__global__ void egkn_zero(float* __restrict__ hs, float* __restrict__ cs,
                          float* __restrict__ cnt, int n_) {
  int t = blockIdx.x * blockDim.x + threadIdx.x;
  if (t < n_ * 32) hs[t] = 0.f;
  if (t < n_ * 3)  cs[t] = 0.f;
  if (t < n_)      cnt[t] = 0.f;
}

// ---------------------------------------------------------------------------
// Fused: k2 = relu(k1@W2+b2); Wk = k2@W3+b3; m = Wk @ h[col];
//        phi = relu(m@cm_w1+cm_b1)@cm_w2; atomics into seg-sum accumulators.
// One wave per 16-edge tile; W3 fragments TDM-staged into LDS, shared by the
// whole block, double buffered. 272 v_wmma per tile.
// ---------------------------------------------------------------------------
__global__ __launch_bounds__(FUSED_WAVES * 32)
void egkn_fused(const _Float16* __restrict__ k1,   // [E,64]
                const _Float16* __restrict__ w2f,  // frag layout
                const float* __restrict__ b2,      // [128]
                const _Float16* __restrict__ w3f,  // frag layout [64][4][32][16]
                const float* __restrict__ b3,      // [1024]
                const float* __restrict__ h,       // [N,32]
                const float* __restrict__ cd,      // [E,3]
                const int* __restrict__ eidx,      // [2,E]
                const float* __restrict__ cmw1,    // [32,32]
                const float* __restrict__ cmb1,    // [32]
                const float* __restrict__ cmw2,    // [32]
                float* __restrict__ h_sum, float* __restrict__ coord_sum,
                float* __restrict__ cnt, int e_, int ntiles) {
  __shared__ __align__(128) _Float16 stage[2][GSZ * 4 * 32 * 16];   // 2 x 16KB
  __shared__ __align__(32)  _Float16 lds_k2[FUSED_WAVES][16][128];  // 16KB
  __shared__ __align__(16)  float    lds_h [FUSED_WAVES][16][32];   // 8KB
  __shared__ __align__(16)  float    lds_m [FUSED_WAVES][16][32];   // 8KB

  const int wave = threadIdx.x >> 5;
  const int lane = threadIdx.x & 31;
  const int tile = blockIdx.x * FUSED_WAVES + wave;
  const bool active = (tile < ntiles);
  const int ebase = active ? tile * 16 : 0;   // inactive waves shadow tile 0,
                                              // keep barriers, skip stores.
  const int half_ = lane >> 4;   // lane group
  const int lrow  = lane & 15;   // A-row (edge) / C-col (N) index
  const int kbase = half_ * 8;

  // ---- kick off TDM for W3 group 0 (overlaps all of stage 1)
  if (wave == 0) {
    unsigned soff = (unsigned)(uintptr_t)(const void*)&stage[0][0];
    tdm_load_1d(w3f, soff, GSZ * 4 * 32 * 16 / 2);   // 4096 dwords
  }

  // ---- A1 fragments: k1 tile [16 edges x 64], f16, two 16x32 fragments
  const _Float16* k1row = k1 + (size_t)(ebase + lrow) * 64;
  h8 a0lo = *(const h8*)(k1row + kbase);
  h8 a0hi = *(const h8*)(k1row + 16 + kbase);
  h8 a1lo = *(const h8*)(k1row + 32 + kbase);
  h8 a1hi = *(const h8*)(k1row + 48 + kbase);
  h16 A1_0 = __builtin_shufflevector(a0lo, a0hi, 0,1,2,3,4,5,6,7,8,9,10,11,12,13,14,15);
  h16 A1_1 = __builtin_shufflevector(a1lo, a1hi, 0,1,2,3,4,5,6,7,8,9,10,11,12,13,14,15);

  // ---- Stage 1: k2 = relu(k1 @ W2 + b2)  (16 WMMAs) -> LDS row-major f16
  #pragma unroll
  for (int nc = 0; nc < 8; ++nc) {
    v8f acc = {};
    h16 B0 = *(const h16*)(w2f + ((size_t)((nc * 2 + 0) * 32 + lane) * 16));
    h16 B1 = *(const h16*)(w2f + ((size_t)((nc * 2 + 1) * 32 + lane) * 16));
    acc = __builtin_amdgcn_wmma_f32_16x16x32_f16(false, A1_0, false, B0,
                                                 (short)0, acc, false, false);
    acc = __builtin_amdgcn_wmma_f32_16x16x32_f16(false, A1_1, false, B1,
                                                 (short)0, acc, false, false);
    float bias = b2[nc * 16 + lrow];
    #pragma unroll
    for (int v = 0; v < 8; ++v) {
      float val = acc[v] + bias;
      lds_k2[wave][v + 8 * half_][nc * 16 + lrow] = (_Float16)(val > 0.f ? val : 0.f);
    }
  }

  // ---- Re-layout C -> A fragments through LDS (4 fragments, K=128)
  h16 A2[4];
  const _Float16* k2row = &lds_k2[wave][lrow][0];
  #pragma unroll
  for (int kf = 0; kf < 4; ++kf) {
    h8 lo = *(const h8*)(k2row + kf * 32 + kbase);
    h8 hi = *(const h8*)(k2row + kf * 32 + 16 + kbase);
    A2[kf] = __builtin_shufflevector(lo, hi, 0,1,2,3,4,5,6,7,8,9,10,11,12,13,14,15);
  }

  // ---- Gather h[col] tile into LDS; zero message accumulator
  {
    const int eh = ebase + (lane >> 1);
    const int colnode = eidx[e_ + eh];
    const float* hp = h + (size_t)colnode * 32 + (lane & 1) * 16;
    float* dsth = &lds_h[wave][lane >> 1][(lane & 1) * 16];
    float* dstm = &lds_m[wave][lane >> 1][(lane & 1) * 16];
    #pragma unroll
    for (int j = 0; j < 16; ++j) { dsth[j] = hp[j]; dstm[j] = 0.f; }
  }

  // ---- Stage 2: 16 TDM-staged groups x GSZ chunks; 4 WMMAs per chunk plus
  //      fused eij,ej->ei contraction via cross-lane butterfly.
  for (int g = 0; g < NGROUPS; ++g) {
    if (wave == 0) __builtin_amdgcn_s_wait_tensorcnt(0);  // group g landed
    __syncthreads();                                      // publish; also
                                                          // retire group g-1
    if (wave == 0 && g + 1 < NGROUPS) {
      unsigned soff = (unsigned)(uintptr_t)(const void*)&stage[(g + 1) & 1][0];
      tdm_load_1d(w3f + (size_t)(g + 1) * (GSZ * 4 * 32 * 16), soff,
                  GSZ * 4 * 32 * 16 / 2);
    }
    const _Float16* sbuf = &stage[g & 1][0];
    #pragma unroll
    for (int lc = 0; lc < GSZ; ++lc) {
      const int c = g * GSZ + lc;
      const _Float16* bp = sbuf + ((size_t)(lc * 4) * 32 + lane) * 16;
      v8f acc = {};
      #pragma unroll
      for (int kf = 0; kf < 4; ++kf) {
        h16 B = *(const h16*)(bp + (size_t)kf * 32 * 16);
        acc = __builtin_amdgcn_wmma_f32_16x16x32_f16(false, A2[kf], false, B,
                                                     (short)0, acc, false, false);
      }
      const float bias = b3[c * 16 + lrow];
      const int i = c >> 1;             // output message dim
      const int jb = (c & 1) << 4;      // h-vector offset within the 32 j's
      float red[8];
      #pragma unroll
      for (int v = 0; v < 8; ++v) {
        int m = v + 8 * half_;          // edge row
        float p = (acc[v] + bias) * lds_h[wave][m][jb + lrow];
        p += __shfl_xor(p, 1, 32);
        p += __shfl_xor(p, 2, 32);
        p += __shfl_xor(p, 4, 32);
        p += __shfl_xor(p, 8, 32);
        red[v] = p;
      }
      if (lrow == 0) {
        #pragma unroll
        for (int v = 0; v < 8; ++v)
          lds_m[wave][v + 8 * half_][i] += red[v];
      }
    }
  }

  // ---- Stage 3: coord head phi + fused segment-sum atomics (2 lanes / edge)
  {
    const int e = ebase + (lane >> 1);
    const int rownode = eidx[e];
    const float* mrow = &lds_m[wave][lane >> 1][0];
    const int dbase = (lane & 1) * 16;
    float partial = 0.f;
    #pragma unroll 2
    for (int d = 0; d < 16; ++d) {
      float s = cmb1[dbase + d];
      #pragma unroll
      for (int j = 0; j < 32; ++j) s += mrow[j] * cmw1[j * 32 + dbase + d];
      partial += (s > 0.f ? s : 0.f) * cmw2[dbase + d];
    }
    float phi = partial + __shfl_xor(partial, 1, 32);
    if (active) {
      #pragma unroll
      for (int j = 0; j < 16; ++j)
        atomicAdd(&h_sum[(size_t)rownode * 32 + dbase + j], mrow[dbase + j]);
      if ((lane & 1) == 0) {
        atomicAdd(&coord_sum[rownode * 3 + 0], cd[(size_t)e * 3 + 0] * phi);
        atomicAdd(&coord_sum[rownode * 3 + 1], cd[(size_t)e * 3 + 1] * phi);
        atomicAdd(&coord_sum[rownode * 3 + 2], cd[(size_t)e * 3 + 2] * phi);
        atomicAdd(&cnt[rownode], 1.0f);
      }
    }
  }
}

// coord += seg_mean(cd*phi); h = relu(h + seg_mean(m))
__global__ void egkn_update(float* __restrict__ h, float* __restrict__ coord,
                            const float* __restrict__ hs, const float* __restrict__ cs,
                            const float* __restrict__ cnt, int n_) {
  int n = blockIdx.x * blockDim.x + threadIdx.x;
  if (n >= n_) return;
  float inv = 1.f / fmaxf(cnt[n], 1.f);
  #pragma unroll
  for (int d = 0; d < 3; ++d) coord[n * 3 + d] += cs[n * 3 + d] * inv;
  #pragma unroll
  for (int d = 0; d < 32; ++d) {
    float v = h[(size_t)n * 32 + d] + hs[(size_t)n * 32 + d] * inv;
    h[(size_t)n * 32 + d] = v > 0.f ? v : 0.f;
  }
}

// out = relu(h@fc2_w1+b1)@fc2_w2 + b2 ; copy coord to output
__global__ void egkn_final(const float* __restrict__ h, const float* __restrict__ coord,
                           const float* __restrict__ w1, const float* __restrict__ b1,
                           const float* __restrict__ w2, const float* __restrict__ b2,
                           float* __restrict__ out, int n_) {
  int n = blockIdx.x * blockDim.x + threadIdx.x;
  if (n >= n_) return;
  float hn[32];
  #pragma unroll
  for (int j = 0; j < 32; ++j) hn[j] = h[(size_t)n * 32 + j];
  float o = b2[0];
  #pragma unroll 4
  for (int d = 0; d < 64; ++d) {
    float s = b1[d];
    #pragma unroll
    for (int j = 0; j < 32; ++j) s += hn[j] * w1[j * 64 + d];
    o += (s > 0.f ? s : 0.f) * w2[d];
  }
  out[n] = o;
  out[n_ + n * 3 + 0] = coord[n * 3 + 0];
  out[n_ + n * 3 + 1] = coord[n * 3 + 1];
  out[n_ + n * 3 + 2] = coord[n * 3 + 2];
}

// ---------------------------------------------------------------------------
extern "C" void kernel_launch(void* const* d_in, const int* in_sizes, int n_in,
                              void* d_out, int out_size, void* d_ws, size_t ws_size,
                              hipStream_t stream) {
  const float* x      = (const float*)d_in[0];
  const int*   eidx   = (const int*)  d_in[1];
  const float* eattr  = (const float*)d_in[2];
  const float* ci     = (const float*)d_in[3];
  const float* fc1_w  = (const float*)d_in[4];
  const float* fc1_b  = (const float*)d_in[5];
  const float* kw1    = (const float*)d_in[6];
  const float* kb1    = (const float*)d_in[7];
  const float* kw2    = (const float*)d_in[8];
  const float* kb2    = (const float*)d_in[9];
  const float* kw3    = (const float*)d_in[10];
  const float* kb3    = (const float*)d_in[11];
  const float* cmw1   = (const float*)d_in[12];
  const float* cmb1   = (const float*)d_in[13];
  const float* cmw2   = (const float*)d_in[14];
  const float* f2w1   = (const float*)d_in[15];
  const float* f2b1   = (const float*)d_in[16];
  const float* f2w2   = (const float*)d_in[17];
  const float* f2b2   = (const float*)d_in[18];

  const int N_ = EGKN_N, E_ = EGKN_E;
  char* ws = (char*)d_ws;
  size_t off = 0;
  auto alloc = [&](size_t bytes) {
    size_t o = off; off = (off + bytes + 255) & ~(size_t)255; return o;
  };
  float*    h     = (float*)   (ws + alloc((size_t)N_ * 32 * 4));
  float*    coord = (float*)   (ws + alloc((size_t)N_ * 3 * 4));
  float*    cd    = (float*)   (ws + alloc((size_t)E_ * 3 * 4));
  _Float16* k1    = (_Float16*)(ws + alloc((size_t)E_ * 64 * 2));
  _Float16* w2f   = (_Float16*)(ws + alloc(8192 * 2));
  _Float16* w3f   = (_Float16*)(ws + alloc(131072 * 2));
  float*    hsum  = (float*)   (ws + alloc((size_t)N_ * 32 * 4));
  float*    csum  = (float*)   (ws + alloc((size_t)N_ * 3 * 4));
  float*    cnt   = (float*)   (ws + alloc((size_t)N_ * 4));
  float*    out   = (float*)d_out;

  egkn_prep<<<512, 256, 0, stream>>>(kw2, kw3, w2f, w3f);
  egkn_init<<<(N_ + 255) / 256, 256, 0, stream>>>(x, fc1_w, fc1_b, ci, h, coord, N_);

  const int ntiles = E_ / 16;                       // 6250
  const int fblocks = (ntiles + FUSED_WAVES - 1) / FUSED_WAVES;
  for (int layer = 0; layer < 3; ++layer) {
    egkn_edge1<<<(E_ + 255) / 256, 256, 0, stream>>>(eidx, eattr, coord, kw1, kb1,
                                                     cd, k1, E_);
    egkn_zero<<<(N_ * 32 + 255) / 256, 256, 0, stream>>>(hsum, csum, cnt, N_);
    egkn_fused<<<fblocks, FUSED_WAVES * 32, 0, stream>>>(
        k1, w2f, kb2, w3f, kb3, h, cd, eidx, cmw1, cmb1, cmw2,
        hsum, csum, cnt, E_, ntiles);
    egkn_update<<<(N_ + 255) / 256, 256, 0, stream>>>(h, coord, hsum, csum, cnt, N_);
  }
  egkn_final<<<(N_ + 255) / 256, 256, 0, stream>>>(h, coord, f2w1, f2b1, f2w2, f2b2,
                                                   out, N_);
}
